// LSTM_30760555774543
// MI455X (gfx1250) — matrix operand order repaired and checked by reading, not compiled, not run on hardware
//
#include <hip/hip_runtime.h>

// CDNA5 / gfx1250 reset-LSTM.
// Phase 0 (parallel): x -> bf16, W_cat = [W_ih | W_hh] -> bf16, bias = b_ih+b_hh,
//                     h_buf0 = bf16(h0), zero per-step barrier counters.
// Phase 1 (persistent, 16 WGs x 256 thr): 1024 sequential steps; per step each
//   wave computes the 4 gate tiles (i,f,g,o) of a 16x16 (batch x j) patch via
//   v_wmma_f32_16x16x32_bf16 over K=1024 ([x_t | keep*h]), applies the LSTM
//   cell in registers (c stays in VGPRs the whole kernel), writes fp32 y and
//   bf16 h, then crosses a device-wide barrier.

typedef __attribute__((ext_vector_type(16))) __bf16 v16bf;
typedef __attribute__((ext_vector_type(8)))  float  v8f;

#define T_STEPS  1024
#define BATCH    64
#define CDIM     512
#define KDIM     1024          // [x | h]
#define NWG      16
#define NTHREADS 256
#define A_STRIDE 1032          // 1024 + 8 pad (ushort elems): conflict-free ds_load_b128

union FragA { v16bf v; uint4 q[2]; };
union FragB { v16bf v; uint4 q[2]; };

__device__ __forceinline__ unsigned short f2bf(float f) {
  union { float f; unsigned u; } x; x.f = f;
  unsigned r = x.u + 0x7FFFu + ((x.u >> 16) & 1u);   // round-to-nearest-even
  return (unsigned short)(r >> 16);
}

__global__ void cvt_x_bf16(const float* __restrict__ x, unsigned short* __restrict__ xb) {
  size_t i = ((size_t)blockIdx.x * NTHREADS + threadIdx.x) * 4;
  float4 v = *(const float4*)(x + i);
  ushort4 o;
  o.x = f2bf(v.x); o.y = f2bf(v.y); o.z = f2bf(v.z); o.w = f2bf(v.w);
  *(ushort4*)(xb + i) = o;
}

__global__ void prep(const float* __restrict__ Wih, const float* __restrict__ Whh,
                     const float* __restrict__ bih, const float* __restrict__ bhh,
                     const float* __restrict__ h0,
                     unsigned short* __restrict__ wcat, float* __restrict__ bias,
                     unsigned short* __restrict__ hb0, unsigned int* __restrict__ counters) {
  int gid = blockIdx.x * NTHREADS + threadIdx.x;     // covers 2048*1024
  int n = gid >> 10, k = gid & 1023;
  float w = (k < CDIM) ? Wih[(size_t)n * CDIM + k] : Whh[(size_t)n * CDIM + (k - CDIM)];
  wcat[gid] = f2bf(w);
  if (gid < 4 * CDIM)      bias[gid] = bih[gid] + bhh[gid];
  if (gid < BATCH * CDIM)  hb0[gid]  = f2bf(h0[gid]);
  if (gid < T_STEPS)       counters[gid] = 0u;       // ws is poisoned; re-init every launch
}

__global__ void __launch_bounds__(NTHREADS, 1)
lstm_scan(const unsigned short* __restrict__ xb,
          const unsigned short* __restrict__ wcat,
          const float* __restrict__ bias,
          const unsigned char* __restrict__ resetv,   // bool8 [T,B]
          const float* __restrict__ c0,
          unsigned short* __restrict__ hb0,
          unsigned short* __restrict__ hb1,
          unsigned int* __restrict__ counters,
          float* __restrict__ y) {
  extern __shared__ unsigned short smemA[];           // [BATCH][A_STRIDE] bf16 bits
  __shared__ float keepf[BATCH];

  const int tid    = threadIdx.x;
  const int lane   = tid & 31;
  const int wave   = tid >> 5;                        // 8 waves
  const int mblock = wave & 3;                        // batch-row block (4 x 16)
  const int jgroup = wave >> 2;                       // 2 j-groups of 16 cols
  const int col    = lane & 15;
  const int rh     = lane >> 4;                       // lane half (ISA VGPR layouts)
  const int jbase  = blockIdx.x * 32 + jgroup * 16;
  const int jcol   = jbase + col;
  const int mrow0  = mblock * 16 + rh * 8;            // C/D elem v -> row mrow0+v
  const int arow   = mblock * 16 + col;               // A-frag row for this lane

  // Per-lane B-operand base pointers (one W_cat row per lane) and bias scalars.
  const unsigned short* wP[4];
  float bs[4];
  #pragma unroll
  for (int g = 0; g < 4; ++g) {
    wP[g] = wcat + ((size_t)(g * CDIM + jcol) * KDIM + rh * 16);
    bs[g] = bias[g * CDIM + jcol];
  }

  // Cell state lives in registers for all 1024 steps (C/D tile layout).
  v8f c;
  #pragma unroll
  for (int v = 0; v < 8; ++v) c[v] = c0[(size_t)(mrow0 + v) * CDIM + jcol];

  const unsigned short* aBase = smemA + (size_t)arow * A_STRIDE + rh * 16;

  for (int t = 0; t < T_STEPS; ++t) {
    const unsigned short* hc = (t & 1) ? hb1 : hb0;
    unsigned short*       hn = (t & 1) ? hb0 : hb1;
    const int t64 = t * BATCH;

    // ---- cooperative LDS fill: A = [x_t | keep * h_prev], 64 x 1024 bf16 ----
    #pragma unroll
    for (int it = 0; it < (BATCH * KDIM / 8) / NTHREADS; ++it) {   // 32 uint4 / thread
      int idx = tid + it * NTHREADS;
      int row = idx >> 7;
      int cc  = (idx & 127) * 8;
      uint4 val;
      if (cc < CDIM) {
        val = *(const uint4*)(xb + ((size_t)(t64 + row) * CDIM + cc));
      } else if (resetv[t64 + row]) {
        val = make_uint4(0u, 0u, 0u, 0u);
      } else {
        val = *(const uint4*)(hc + ((size_t)row * CDIM + (cc - CDIM)));
      }
      *(uint4*)(smemA + (size_t)row * A_STRIDE + cc) = val;
    }
    if (tid < BATCH) keepf[tid] = resetv[t64 + tid] ? 0.0f : 1.0f;
    __syncthreads();

    // reset carried cell state for reset batch rows
    #pragma unroll
    for (int v = 0; v < 8; ++v) c[v] *= keepf[mrow0 + v];

    // ---- gates = [x_t | h] . W_cat^T : 4 tiles (i,f,g,o), K=1024 ----
    v8f aI = {}, aF = {}, aG = {}, aO = {};
    #pragma unroll 4
    for (int ks = 0; ks < KDIM / 32; ++ks) {
      const int k0 = ks * 32;
      FragA a;
      a.q[0] = *(const uint4*)(aBase + k0);        // K = k0 + rh*16 + 0..7
      a.q[1] = *(const uint4*)(aBase + k0 + 8);    // K = k0 + rh*16 + 8..15
      FragB b0, b1, b2, b3;                        // K = k0 + rh*16 + 0..15
      b0.q[0] = *(const uint4*)(wP[0] + k0); b0.q[1] = *(const uint4*)(wP[0] + k0 + 8);
      b1.q[0] = *(const uint4*)(wP[1] + k0); b1.q[1] = *(const uint4*)(wP[1] + k0 + 8);
      b2.q[0] = *(const uint4*)(wP[2] + k0); b2.q[1] = *(const uint4*)(wP[2] + k0 + 8);
      b3.q[0] = *(const uint4*)(wP[3] + k0); b3.q[1] = *(const uint4*)(wP[3] + k0 + 8);
      aI = __builtin_amdgcn_wmma_f32_16x16x32_bf16(false, a.v, false, b0.v, (short)0, aI, false, false);
      aF = __builtin_amdgcn_wmma_f32_16x16x32_bf16(false, a.v, false, b1.v, (short)0, aF, false, false);
      aG = __builtin_amdgcn_wmma_f32_16x16x32_bf16(false, a.v, false, b2.v, (short)0, aG, false, false);
      aO = __builtin_amdgcn_wmma_f32_16x16x32_bf16(false, a.v, false, b3.v, (short)0, aO, false, false);
    }

    // ---- LSTM cell (torch gate order i,f,g,o), fp32 ----
    #pragma unroll
    for (int v = 0; v < 8; ++v) {
      float iv = aI[v] + bs[0];
      float fv = aF[v] + bs[1];
      float gv = aG[v] + bs[2];
      float ov = aO[v] + bs[3];
      float ig = 1.0f / (1.0f + __expf(-iv));
      float fg = 1.0f / (1.0f + __expf(-fv));
      float og = 1.0f / (1.0f + __expf(-ov));
      float cv = fg * c[v] + ig * tanhf(gv);
      c[v] = cv;
      float hv = og * tanhf(cv);
      int row = mrow0 + v;
      y[((size_t)t64 + row) * CDIM + jcol] = hv;       // fp32 output
      hn[(size_t)row * CDIM + jcol] = f2bf(hv);        // bf16 operand for next step
    }

    // ---- device-wide barrier: publish h to all 16 workgroups ----
    if (t + 1 < T_STEPS) {
      __threadfence();
      __syncthreads();
      if (tid == 0) {
        __hip_atomic_fetch_add(&counters[t], 1u, __ATOMIC_ACQ_REL, __HIP_MEMORY_SCOPE_AGENT);
        while (__hip_atomic_load(&counters[t], __ATOMIC_ACQUIRE, __HIP_MEMORY_SCOPE_AGENT) < NWG) {
          __builtin_amdgcn_s_sleep(1);
        }
      }
      __syncthreads();
    }
  }
}

extern "C" void kernel_launch(void* const* d_in, const int* in_sizes, int n_in,
                              void* d_out, int out_size, void* d_ws, size_t ws_size,
                              hipStream_t stream) {
  (void)in_sizes; (void)n_in; (void)out_size; (void)ws_size;
  const float*         x     = (const float*)d_in[0];
  const unsigned char* reset = (const unsigned char*)d_in[1];   // jax bool -> byte mask
  const float*         h0    = (const float*)d_in[2];
  const float*         c0    = (const float*)d_in[3];
  const float*         Wih   = (const float*)d_in[4];
  const float*         Whh   = (const float*)d_in[5];
  const float*         bih   = (const float*)d_in[6];
  const float*         bhh   = (const float*)d_in[7];
  float* y = (float*)d_out;

  char* ws = (char*)d_ws;
  size_t off = 0;
  auto alloc = [&](size_t bytes) {
    char* p = ws + off;
    off += (bytes + 255) & ~(size_t)255;
    return p;
  };
  unsigned short* xb       = (unsigned short*)alloc((size_t)T_STEPS * BATCH * CDIM * 2); // 64 MB
  unsigned short* wcat     = (unsigned short*)alloc((size_t)4 * CDIM * KDIM * 2);        // 4 MB
  float*          bias     = (float*)alloc((size_t)4 * CDIM * 4);
  unsigned short* hb0      = (unsigned short*)alloc((size_t)BATCH * CDIM * 2);
  unsigned short* hb1      = (unsigned short*)alloc((size_t)BATCH * CDIM * 2);
  unsigned int*   counters = (unsigned int*)alloc((size_t)T_STEPS * 4);

  cvt_x_bf16<<<(T_STEPS * BATCH * CDIM) / (NTHREADS * 4), NTHREADS, 0, stream>>>(x, xb);
  prep<<<(4 * CDIM * KDIM) / NTHREADS, NTHREADS, 0, stream>>>(Wih, Whh, bih, bhh, h0,
                                                              wcat, bias, hb0, counters);
  size_t smem = (size_t)BATCH * A_STRIDE * 2;   // 132,096 B dynamic LDS (WGP has 320 KB)
  lstm_scan<<<NWG, NTHREADS, smem, stream>>>(xb, wcat, bias, reset, c0, hb0, hb1, counters, y);
}
